// EdgeFeatures_18047452578373
// MI455X (gfx1250) — matrix operand order repaired and checked by reading, not compiled, not run on hardware
//
#include <hip/hip_runtime.h>
#include <hip/hip_bf16.h>

typedef _Float16 v16h __attribute__((ext_vector_type(16)));
typedef _Float16 v8h  __attribute__((ext_vector_type(8)));
typedef _Float16 v4h  __attribute__((ext_vector_type(4)));
typedef float    v8f  __attribute__((ext_vector_type(8)));

#define C_IN  128
#define C_HID 256
#define C_OUT 128

// ---------------------------------------------------------------------------
// Kernel 1: pack layer-1/layer-2 weights (f32 row-major) into f16 WMMA
// B-fragment layout: [n_tile][k_tile][lane 0..31][16 halfs]
//   lane l (<16)  = column n_tile*16+l, K = k*32 + 0..15
//   lane l (>=16) = column n_tile*16+(l-16), K = k*32 + 16..31
// ---------------------------------------------------------------------------
__global__ void pack_weights_kernel(const float* __restrict__ W1a,
                                    const float* __restrict__ W1b,
                                    const float* __restrict__ W2a,
                                    const float* __restrict__ W2b,
                                    _Float16* __restrict__ pB1a,
                                    _Float16* __restrict__ pB1b,
                                    _Float16* __restrict__ pB2a,
                                    _Float16* __restrict__ pB2b) {
    int gid = blockIdx.x * blockDim.x + threadIdx.x;   // 0 .. 131071
    int m = gid >> 15;            // which matrix
    int r = gid & 32767;          // element within packed matrix (32768 halfs)
    int j    = r & 15;
    int lane = (r >> 4) & 31;
    int colInTile = lane & 15;
    int khalf = (lane >> 4) * 16;
    if (m < 2) {
        // layer 1: [16 n][4 k] tiles, W1 is [256][128]
        int n = r >> 11;
        int k = (r >> 9) & 3;
        const float* W = (m == 0) ? W1a : W1b;
        _Float16*    P = (m == 0) ? pB1a : pB1b;
        int row  = n * 16 + colInTile;        // hidden channel (B column)
        int colk = k * 32 + khalf + j;        // input channel (K)
        P[r] = (_Float16)W[row * C_IN + colk];
    } else {
        // layer 2: [8 n][8 k] tiles, W2 is [128][256]
        int n = r >> 12;
        int k = (r >> 9) & 7;
        const float* W = (m == 2) ? W2a : W2b;
        _Float16*    P = (m == 2) ? pB2a : pB2b;
        int row  = n * 16 + colInTile;        // output channel (B column)
        int colk = k * 32 + khalf + j;        // hidden channel (K)
        P[r] = (_Float16)W[row * C_HID + colk];
    }
}

// ---------------------------------------------------------------------------
// Kernel 2: global-feature FCNN (c) + fold in b2a + b2b -> gvec[128] (f32)
// ---------------------------------------------------------------------------
__global__ void global_branch_kernel(const float* __restrict__ g,
                                     const float* __restrict__ W1c,
                                     const float* __restrict__ b1c,
                                     const float* __restrict__ W2c,
                                     const float* __restrict__ b2c,
                                     const float* __restrict__ b2a,
                                     const float* __restrict__ b2b,
                                     float* __restrict__ gvec) {
    __shared__ float hidc[C_HID];
    int t = threadIdx.x;
    float a = b1c[t];
    #pragma unroll 8
    for (int i = 0; i < C_IN; ++i) a += g[i] * W1c[t * C_IN + i];
    hidc[t] = a > 0.f ? a : 0.f;
    __syncthreads();
    if (t < C_OUT) {
        float o = b2c[t] + b2a[t] + b2b[t];
        #pragma unroll 8
        for (int j = 0; j < C_HID; ++j) o += hidc[j] * W2c[t * C_HID + j];
        gvec[t] = o;
    }
}

// ---------------------------------------------------------------------------
// Main fused kernel: 4 waves/block, 16 edges/wave
// ---------------------------------------------------------------------------
// A-fragment (16x32 f16) from row-major LDS tile.
// Lane l<16 = row l, K k*32+0..15; lane l>=16 = row l-16,
// K k*32+16..31  (contiguous order, matches packed B).
__device__ __forceinline__ v16h load_a_frag(const _Float16* base, int stride,
                                            int k, int lane) {
    int row = lane & 15;
    int hi  = lane >> 4;
    const _Float16* q = base + row * stride + k * 32 + hi * 16;
    v8h lo = *(const v8h*)q;
    v8h hh = *(const v8h*)(q + 8);
    return __builtin_shufflevector(lo, hh, 0,1,2,3,4,5,6,7,8,9,10,11,12,13,14,15);
}

__device__ __forceinline__ v16h load_b_frag(const _Float16* pack, int idx,
                                            int lane) {
    const _Float16* q = pack + (size_t)idx * 512 + lane * 16;
    v8h lo = *(const v8h*)q;
    v8h hh = *(const v8h*)(q + 8);
    return __builtin_shufflevector(lo, hh, 0,1,2,3,4,5,6,7,8,9,10,11,12,13,14,15);
}

__global__ __launch_bounds__(128)
void edge_main_kernel(const float* __restrict__ nf,
                      const long long* __restrict__ eidx,
                      const float* __restrict__ ef,
                      const float* __restrict__ b1a,
                      const float* __restrict__ b1b,
                      const _Float16* __restrict__ pB1a,
                      const _Float16* __restrict__ pB1b,
                      const _Float16* __restrict__ pB2a,
                      const _Float16* __restrict__ pB2b,
                      const float* __restrict__ gvec,
                      float* __restrict__ out,
                      int E) {
    // per-wave LDS slice (8192 halfs = 16KB):
    //   [0,2048)     node_sum 16x128 f16
    //   [2048,4096)  edge     16x128 f16
    //   [4096,8192)  hidden   16x256 f16
    __shared__ __align__(16) _Float16 lds[4 * 8192];

    const int tid  = threadIdx.x;
    const int wave = tid >> 5;
    const int lane = tid & 31;
    const int ebase = blockIdx.x * 64 + wave * 16;

    _Float16* ndA = &lds[wave * 8192];
    _Float16* egA = ndA + 2048;
    _Float16* hid = ndA + 4096;

    // ---------------- stage inputs: gather node_sum + edge feats (f32->f16)
    {
        int row = lane >> 1;
        int c0  = (lane & 1) * 64;
        long long e = (long long)ebase + row;
        if (e >= E) e = 0;                       // clamp tail (rows not stored)
        long long s = eidx[e];
        long long d = eidx[(long long)E + e];
        const float4* ps = (const float4*)(nf + s * C_IN + c0);
        const float4* pd = (const float4*)(nf + d * C_IN + c0);
        const float4* pe = (const float4*)(ef + e * C_IN + c0);
        _Float16* nrow = ndA + row * 128 + c0;
        _Float16* erow = egA + row * 128 + c0;
        #pragma unroll
        for (int i = 0; i < 16; ++i) {
            float4 a = ps[i], b = pd[i], gg = pe[i];
            v4h hn = { (_Float16)(a.x + b.x), (_Float16)(a.y + b.y),
                       (_Float16)(a.z + b.z), (_Float16)(a.w + b.w) };
            v4h he = { (_Float16)gg.x, (_Float16)gg.y,
                       (_Float16)gg.z, (_Float16)gg.w };
            *(v4h*)(nrow + i * 4) = hn;
            *(v4h*)(erow + i * 4) = he;
        }
    }
    __syncthreads();

    const v8f vzero = {0.f,0.f,0.f,0.f,0.f,0.f,0.f,0.f};
    const int col = lane & 15;
    const int hi2 = lane >> 4;

    v8f oacc[8];
    #pragma unroll
    for (int n = 0; n < 8; ++n) oacc[n] = vzero;

    // ---------------- two branches: (node_sum, FCNN a) then (edge, FCNN b)
    for (int br = 0; br < 2; ++br) {
        const _Float16* Asrc  = br ? egA  : ndA;
        const _Float16* pb1   = br ? pB1b : pB1a;
        const _Float16* pb2   = br ? pB2b : pB2a;
        const float*    bias1 = br ? b1b  : b1a;

        // layer 1: hidden[16x256] in two halves of 8 n-tiles
        for (int h = 0; h < 2; ++h) {
            v8f acc[8];
            #pragma unroll
            for (int n = 0; n < 8; ++n) acc[n] = vzero;
            #pragma unroll
            for (int k = 0; k < 4; ++k) {
                v16h a = load_a_frag(Asrc, 128, k, lane);
                #pragma unroll
                for (int n = 0; n < 8; ++n) {
                    v16h b = load_b_frag(pb1, (h * 8 + n) * 4 + k, lane);
                    acc[n] = __builtin_amdgcn_wmma_f32_16x16x32_f16(
                        false, a, false, b, (short)0, acc[n], false, false);
                }
            }
            // bias + relu, repack hidden to LDS (f16)
            #pragma unroll
            for (int n = 0; n < 8; ++n) {
                int nc = (h * 8 + n) * 16 + col;
                float bv = bias1[nc];
                #pragma unroll
                for (int r = 0; r < 8; ++r) {
                    float v = acc[n][r] + bv;
                    hid[(r + hi2 * 8) * 256 + nc] = (_Float16)(v > 0.f ? v : 0.f);
                }
            }
        }
        __syncthreads();

        // layer 2: out += relu(hidden) @ W2^T
        #pragma unroll
        for (int k = 0; k < 8; ++k) {
            v16h a = load_a_frag(hid, 256, k, lane);
            #pragma unroll
            for (int n = 0; n < 8; ++n) {
                v16h b = load_b_frag(pb2, n * 8 + k, lane);
                oacc[n] = __builtin_amdgcn_wmma_f32_16x16x32_f16(
                    false, a, false, b, (short)0, oacc[n], false, false);
            }
        }
        __syncthreads();
    }

    // ---------------- epilogue: +gvec, instance-norm per row, dropout,
    //                  relu, residual, store
    float gv[8];
    #pragma unroll
    for (int n = 0; n < 8; ++n) gv[n] = gvec[n * 16 + col];

    float mean[8], rstd[8];
    #pragma unroll
    for (int r = 0; r < 8; ++r) {
        float s1 = 0.f, s2 = 0.f;
        #pragma unroll
        for (int n = 0; n < 8; ++n) {
            float v = oacc[n][r] + gv[n];
            s1 += v;
            s2 += v * v;
        }
        // reduce within 16-lane half-groups (rows r and r+8 independently)
        #pragma unroll
        for (int m = 8; m >= 1; m >>= 1) {
            s1 += __shfl_xor(s1, m, 32);
            s2 += __shfl_xor(s2, m, 32);
        }
        float mu  = s1 * (1.0f / 128.0f);
        float var = s2 * (1.0f / 128.0f) - mu * mu;
        mean[r] = mu;
        rstd[r] = rsqrtf(var + 1e-5f);
    }

    const unsigned keep_thresh = 15099494u;   // 0.9 * 2^24
    #pragma unroll
    for (int n = 0; n < 8; ++n) {
        int c = n * 16 + col;
        #pragma unroll
        for (int r = 0; r < 8; ++r) {
            int row = r + hi2 * 8;
            long long e = (long long)ebase + row;
            if (e < E) {
                float v = (oacc[n][r] + gv[n] - mean[r]) * rstd[r];
                // deterministic hash-based dropout (p = 0.1)
                unsigned h = (unsigned)(e * 128 + c);
                h ^= h >> 16; h *= 0x7feb352du;
                h ^= h >> 15; h *= 0x846ca68bu;
                h ^= h >> 16;
                bool keep = (h & 0xFFFFFFu) < keep_thresh;
                v = keep ? v * (1.0f / 0.9f) : 0.0f;
                float res = ef[e * C_IN + c] + (v > 0.f ? v : 0.f);
                out[e * C_IN + c] = res;
            }
        }
    }
}

// ---------------------------------------------------------------------------
extern "C" void kernel_launch(void* const* d_in, const int* in_sizes, int n_in,
                              void* d_out, int out_size, void* d_ws, size_t ws_size,
                              hipStream_t stream) {
    const float*     nf   = (const float*)d_in[0];
    const long long* eidx = (const long long*)d_in[1];
    const float*     ef   = (const float*)d_in[2];
    const float*     g    = (const float*)d_in[3];
    const float* W1a = (const float*)d_in[4];
    const float* b1a = (const float*)d_in[5];
    const float* W2a = (const float*)d_in[6];
    const float* b2a = (const float*)d_in[7];
    const float* W1b = (const float*)d_in[8];
    const float* b1b = (const float*)d_in[9];
    const float* W2b = (const float*)d_in[10];
    const float* b2b = (const float*)d_in[11];
    const float* W1c = (const float*)d_in[12];
    const float* b1c = (const float*)d_in[13];
    const float* W2c = (const float*)d_in[14];
    const float* b2c = (const float*)d_in[15];

    int E = in_sizes[2] / C_IN;

    char* ws = (char*)d_ws;
    float*     gvec = (float*)ws;                                // 512 B
    _Float16*  pB1a = (_Float16*)(ws + 1024);                    // 64 KB each
    _Float16*  pB1b = (_Float16*)(ws + 1024 + 1 * 65536);
    _Float16*  pB2a = (_Float16*)(ws + 1024 + 2 * 65536);
    _Float16*  pB2b = (_Float16*)(ws + 1024 + 3 * 65536);

    pack_weights_kernel<<<512, 256, 0, stream>>>(W1a, W1b, W2a, W2b,
                                                 pB1a, pB1b, pB2a, pB2b);
    global_branch_kernel<<<1, 256, 0, stream>>>(g, W1c, b1c, W2c, b2c,
                                                b2a, b2b, gvec);
    int blocks = (E + 63) / 64;
    edge_main_kernel<<<blocks, 128, 0, stream>>>(nf, eidx, ef, b1a, b1b,
                                                 pB1a, pB1b, pB2a, pB2b,
                                                 gvec, (float*)d_out, E);
}
